// BioDSNN_Model_72275709657428
// MI455X (gfx1250) — matrix operand order repaired and checked by reading, not compiled, not run on hardware
//
#include <hip/hip_runtime.h>
#include <math.h>

// ---------------------------------------------------------------------------
// CDNA5 (gfx1250) implementation of the BioDSNN reference model.
// All large matmuls use v_wmma_f32_16x16x32_f16 (wave32 WMMA, f32 accumulate).
// Attention is fused flash-style; K/V chunk tiles are staged into LDS with the
// Tensor Data Mover (tensor_load_to_lds + s_wait_tensorcnt) when available,
// double-buffered so the next chunk's DMA overlaps the current chunk's WMMAs.
// ---------------------------------------------------------------------------

typedef __attribute__((ext_vector_type(16))) _Float16 v16h;
typedef __attribute__((ext_vector_type(8)))  _Float16 v8h;
typedef __attribute__((ext_vector_type(8)))  float    v8f;
typedef __attribute__((ext_vector_type(4)))  unsigned int v4u;
typedef __attribute__((ext_vector_type(8)))  int      v8i;
typedef __attribute__((ext_vector_type(4)))  int      v4i;

#define G_ 4096
#define H_ 64
#define B_ 4
#define HD_ 32

#if __has_builtin(__builtin_amdgcn_tensor_load_to_lds) && __has_builtin(__builtin_amdgcn_s_wait_tensorcnt)
#define USE_TDM 1
#else
#define USE_TDM 0
#endif

#if USE_TDM
// Issue a TDM load of a 32x32 f16 tile (row stride 32 elements) from global
// memory into LDS at byte offset lds_off.  D# layout per CDNA5 ISA §8.3/8.4:
//   group0: count=1 | lds_addr | global_addr[56:0] | type=2
//   group1: data_size=1(2B), tensor_dim0=32, tensor_dim1=4096,
//           tile_dim0=32, tile_dim1=32, tensor_dim0_stride=32
__device__ __forceinline__ void tdm_load_tile_32x32(const _Float16* gsrc,
                                                    unsigned lds_off) {
  unsigned long long ga = (unsigned long long)(uintptr_t)gsrc;
  v4u g0;
  g0[0] = 1u;                                            // count=1 (valid user D#)
  g0[1] = lds_off;                                       // lds_addr (bytes)
  g0[2] = (unsigned)(ga & 0xffffffffu);                  // global_addr[31:0]
  g0[3] = (unsigned)((ga >> 32) & 0x01ffffffu) | (2u << 30); // addr[56:32] | type=2
  v8i g1;
  g1[0] = (int)(1u << 16);        // workgroup_mask=0, data_size=1 (2 bytes)
  g1[1] = (int)(32u << 16);       // tensor_dim0 = 32 (bits 79:48, low half)
  g1[2] = (int)(4096u << 16);     // tensor_dim0 hi=0, tensor_dim1 = 4096 (low 16)
  g1[3] = (int)(32u << 16);       // tensor_dim1 hi=0, tile_dim0 = 32
  g1[4] = 32;                     // tile_dim1 = 32, tile_dim2 = 0
  g1[5] = 32;                     // tensor_dim0_stride = 32 (low 32 of 48)
  g1[6] = 0;                      // stride hi, tensor_dim1_stride lo
  g1[7] = 0;
  v4i z = {0, 0, 0, 0};
#if defined(__clang_major__) && (__clang_major__ >= 23)
  v8i z8 = {0, 0, 0, 0, 0, 0, 0, 0};
  __builtin_amdgcn_tensor_load_to_lds(g0, g1, z, z, z8, 0);
#else
  __builtin_amdgcn_tensor_load_to_lds(g0, g1, z, z, 0);
#endif
}
#endif

// ---------------------------------------------------------------------------
// Generic WMMA GEMM: C[M,N] = A[M,K] @ W[N,K]^T + bias, optional ReLU,
// optional f16 mirror of the output.  A, W are f16; C is f32.
// One wave computes one 16x16 tile.  K must be a multiple of 32.
//
// Operand layouts used here (contiguous k order, not the ISA table order):
//  A (16x32 f16): lane<16 -> row=lane,   halfs = K[kt+0..15]
//                 lane>=16 -> row=lane-16, halfs = K[kt+16..31]
//  B (32x16 f16): lane<16 -> col=lane,   halfs = K[kt+0..15]
//                 lane>=16 -> col=lane-16, halfs = K[kt+16..31]
//  C (16x16 f32): VGPR i, lane<16 -> (row=i,  col=lane)
//                          lane>=16 -> (row=i+8,col=lane-16)
// ---------------------------------------------------------------------------
__global__ void k_gemm_wmma(const _Float16* __restrict__ A,
                            const _Float16* __restrict__ W,
                            const float* __restrict__ bias,
                            float* __restrict__ C,
                            _Float16* __restrict__ Ch,
                            int M, int N, int K, int act) {
  int lane = threadIdx.x & 31;
  int tn = blockIdx.x, tm = blockIdx.y;
  int n    = tn * 16 + (lane & 15);
  int rowA = tm * 16 + (lane & 15);
  const _Float16* arow = A + (size_t)rowA * K + ((lane & 16) ? 16 : 0);
  const _Float16* wrow = W + (size_t)n * K + ((lane & 16) ? 16 : 0);

  v8f acc = {};
  for (int kt = 0; kt < K; kt += 32) {
    v8h a0 = *(const v8h*)(arow + kt);
    v8h a1 = *(const v8h*)(arow + kt + 8);
    v8h b0 = *(const v8h*)(wrow + kt);
    v8h b1 = *(const v8h*)(wrow + kt + 8);
    v16h av, bv;
#pragma unroll
    for (int e = 0; e < 8; e++) { av[e] = a0[e]; av[e + 8] = a1[e];
                                  bv[e] = b0[e]; bv[e + 8] = b1[e]; }
    acc = __builtin_amdgcn_wmma_f32_16x16x32_f16(false, av, false, bv,
                                                 (short)0, acc, false, false);
  }
  float bn = bias[n];
  int rbase = tm * 16 + ((lane & 16) ? 8 : 0);
#pragma unroll
  for (int i = 0; i < 8; i++) {
    float v = acc[i] + bn;
    if (act) v = fmaxf(v, 0.f);
    size_t off = (size_t)(rbase + i) * N + n;
    C[off] = v;
    if (Ch) Ch[off] = (_Float16)v;
  }
}

// ---------------------------------------------------------------------------
// Flash attention: one wave per (batch*head, 16-query tile).
// Q/K/V are f16 [B*NH][G][32].  mask f32 [G][G].  ctx f32 [B][G][64].
// ---------------------------------------------------------------------------
__global__ void k_flash(const _Float16* __restrict__ qh,
                        const _Float16* __restrict__ kh,
                        const _Float16* __restrict__ vh,
                        const float* __restrict__ mask,
                        float* __restrict__ ctx, float scale) {
  __shared__ __align__(16) _Float16 Psh[16 * 32];
#if USE_TDM
  __shared__ __align__(16) _Float16 Ksh[2][32 * 32];
  __shared__ __align__(16) _Float16 Vsh[2][32 * 32];
#endif
  int lane = threadIdx.x & 31;
  int qt = blockIdx.x;
  int bh = blockIdx.y;
  int b = bh >> 1, h = bh & 1;
  const _Float16* Q  = qh + (size_t)bh * G_ * HD_;
  const _Float16* Kp = kh + (size_t)bh * G_ * HD_;
  const _Float16* Vp = vh + (size_t)bh * G_ * HD_;

  // Q tile (A operand), loaded once
  int qrow = qt * 16 + (lane & 15);
  const _Float16* qp = Q + (size_t)qrow * HD_ + ((lane & 16) ? 16 : 0);
  v8h q0 = *(const v8h*)qp, q1 = *(const v8h*)(qp + 8);
  v16h qa;
#pragma unroll
  for (int e = 0; e < 8; e++) { qa[e] = q0[e]; qa[e + 8] = q1[e]; }

  v8f o0 = {}, o1 = {};
  float m[8], l[8];
#pragma unroll
  for (int i = 0; i < 8; i++) { m[i] = -3.0e38f; l[i] = 0.f; }

#if USE_TDM
  // prologue: DMA first K/V chunk into buffer 0
  tdm_load_tile_32x32(Kp, (unsigned)(uintptr_t)&Ksh[0][0]);
  tdm_load_tile_32x32(Vp, (unsigned)(uintptr_t)&Vsh[0][0]);
#endif

  for (int kc = 0; kc < G_; kc += 32) {
#if USE_TDM
    int buf = (kc >> 5) & 1;
    if (kc + 32 < G_) {
      // issue next chunk's DMAs, then wait for the current chunk only
      tdm_load_tile_32x32(Kp + (size_t)(kc + 32) * HD_,
                          (unsigned)(uintptr_t)&Ksh[buf ^ 1][0]);
      tdm_load_tile_32x32(Vp + (size_t)(kc + 32) * HD_,
                          (unsigned)(uintptr_t)&Vsh[buf ^ 1][0]);
      __builtin_amdgcn_s_wait_tensorcnt(2);
    } else {
      __builtin_amdgcn_s_wait_tensorcnt(0);
    }
    __syncthreads();
    const _Float16* kp0 = &Ksh[buf][(lane & 15) * HD_ + ((lane & 16) ? 16 : 0)];
    const _Float16* kp1 = kp0 + 16 * HD_;
#else
    if (kc + 32 < G_) __builtin_prefetch(Kp + (size_t)(kc + 32) * HD_, 0, 1);
    int key0 = kc + (lane & 15);
    const _Float16* kp0 = Kp + (size_t)key0 * HD_ + ((lane & 16) ? 16 : 0);
    const _Float16* kp1 = Kp + (size_t)(key0 + 16) * HD_ + ((lane & 16) ? 16 : 0);
#endif
    // --- scores: two WMMAs (keys kc..kc+15 and kc+16..kc+31)
    v8h t0 = *(const v8h*)kp0, t1 = *(const v8h*)(kp0 + 8);
    v8h u0 = *(const v8h*)kp1, u1 = *(const v8h*)(kp1 + 8);
    v16h bk0, bk1;
#pragma unroll
    for (int e = 0; e < 8; e++) { bk0[e] = t0[e]; bk0[e + 8] = t1[e];
                                  bk1[e] = u0[e]; bk1[e + 8] = u1[e]; }
    v8f s0 = {}, s1 = {};
    s0 = __builtin_amdgcn_wmma_f32_16x16x32_f16(false, qa, false, bk0, (short)0, s0, false, false);
    s1 = __builtin_amdgcn_wmma_f32_16x16x32_f16(false, qa, false, bk1, (short)0, s1, false, false);

    // --- online softmax (per-row reductions within each 16-lane group)
#pragma unroll
    for (int i = 0; i < 8; i++) {
      int r = qt * 16 + i + ((lane & 16) ? 8 : 0);
      float v0 = s0[i] * scale + mask[(size_t)r * G_ + kc + (lane & 15)];
      float v1 = s1[i] * scale + mask[(size_t)r * G_ + kc + 16 + (lane & 15)];
      float mx = fmaxf(v0, v1);
      mx = fmaxf(mx, __shfl_xor(mx, 1, 32));
      mx = fmaxf(mx, __shfl_xor(mx, 2, 32));
      mx = fmaxf(mx, __shfl_xor(mx, 4, 32));
      mx = fmaxf(mx, __shfl_xor(mx, 8, 32));
      float mnew = fmaxf(m[i], mx);
      float sc = __expf(m[i] - mnew);
      float p0 = __expf(v0 - mnew), p1 = __expf(v1 - mnew);
      float rs = p0 + p1;
      rs += __shfl_xor(rs, 1, 32);
      rs += __shfl_xor(rs, 2, 32);
      rs += __shfl_xor(rs, 4, 32);
      rs += __shfl_xor(rs, 8, 32);
      l[i] = l[i] * sc + rs;
      m[i] = mnew;
      o0[i] *= sc; o1[i] *= sc;
      s0[i] = p0; s1[i] = p1;
    }

    // --- convert P from C-layout to A-layout via LDS
    __syncthreads();
#pragma unroll
    for (int i = 0; i < 8; i++) {
      int rl = i + ((lane & 16) ? 8 : 0);
      Psh[rl * 32 + (lane & 15)]      = (_Float16)s0[i];
      Psh[rl * 32 + 16 + (lane & 15)] = (_Float16)s1[i];
    }
    __syncthreads();
    const _Float16* pp = &Psh[(lane & 15) * 32 + ((lane & 16) ? 16 : 0)];
    v16h pa;
#pragma unroll
    for (int e = 0; e < 8; e++) { pa[e] = pp[e]; pa[e + 8] = pp[8 + e]; }

    // --- O += P x V  (B operand: V[key][dim], transposed gather)
#if USE_TDM
    const _Float16* vc = &Vsh[buf][(((lane & 16) ? 16 : 0)) * HD_ + (lane & 15)];
#else
    const _Float16* vc = Vp + (size_t)(kc + ((lane & 16) ? 16 : 0)) * HD_ + (lane & 15);
#endif
    v16h vb0, vb1;
#pragma unroll
    for (int e = 0; e < 16; e++) { vb0[e] = vc[e * HD_]; vb1[e] = vc[e * HD_ + 16]; }
    o0 = __builtin_amdgcn_wmma_f32_16x16x32_f16(false, pa, false, vb0, (short)0, o0, false, false);
    o1 = __builtin_amdgcn_wmma_f32_16x16x32_f16(false, pa, false, vb1, (short)0, o1, false, false);
  }

  // --- epilogue: normalize and store (heads interleaved into H=64)
#pragma unroll
  for (int i = 0; i < 8; i++) {
    int r = qt * 16 + i + ((lane & 16) ? 8 : 0);
    float inv = 1.f / l[i];
    size_t off = ((size_t)b * G_ + r) * H_ + h * HD_ + (lane & 15);
    ctx[off] = o0[i] * inv;
    ctx[off + 16] = o1[i] * inv;
  }
}

// ------------------------------- small kernels -----------------------------
__global__ void k_fill(float* p, int n, float v) {
  int i = blockIdx.x * blockDim.x + threadIdx.x;
  if (i < n) p[i] = v;
}
__global__ void k_cast16(const float* __restrict__ X, _Float16* __restrict__ Y, int n) {
  int i = blockIdx.x * blockDim.x + threadIdx.x;
  if (i < n) Y[i] = (_Float16)X[i];
}
__global__ void k_edge_deg(const int* __restrict__ src, const int* __restrict__ dst,
                           const float* __restrict__ w, float* __restrict__ deg, int E) {
  int e = blockIdx.x * blockDim.x + threadIdx.x;
  if (e < E) atomicAdd(&deg[dst[e]], w[e]);
  (void)src;
}
__global__ void k_dinv(const float* __restrict__ deg, float* __restrict__ dinv, int n) {
  int i = blockIdx.x * blockDim.x + threadIdx.x;
  if (i < n) { float d = deg[i]; dinv[i] = (d > 0.f) ? rsqrtf(d) : 0.f; }
}
__global__ void k_sg_self(const float* __restrict__ x, const float* __restrict__ dinv,
                          float* __restrict__ agg, int n) {   // n = N*64
  int i = blockIdx.x * blockDim.x + threadIdx.x;
  if (i < n) { int node = i >> 6; float d = dinv[node]; agg[i] = d * d * x[i]; }
}
__global__ void k_sg_edge(const int* __restrict__ src, const int* __restrict__ dst,
                          const float* __restrict__ w, const float* __restrict__ dinv,
                          const float* __restrict__ x, float* __restrict__ agg, int E) {
  int idx = blockIdx.x * blockDim.x + threadIdx.x;
  if (idx >= E * 64) return;
  int e = idx >> 6, f = idx & 63;
  int s = src[e], d = dst[e];
  float norm = dinv[d] * w[e] * dinv[s];
  atomicAdd(&agg[d * 64 + f], norm * x[s * 64 + f]);
}
__global__ void k_bn_stats(const float* __restrict__ X, int M, int F,
                           float* __restrict__ stats, float eps) {
  int f = blockIdx.x;
  float s = 0.f, ss = 0.f;
  for (int r = threadIdx.x; r < M; r += blockDim.x) {
    float v = X[(size_t)r * F + f]; s += v; ss += v * v;
  }
  __shared__ float shs[256], shq[256];
  shs[threadIdx.x] = s; shq[threadIdx.x] = ss;
  __syncthreads();
  for (int st = 128; st > 0; st >>= 1) {
    if ((int)threadIdx.x < st) { shs[threadIdx.x] += shs[threadIdx.x + st];
                                 shq[threadIdx.x] += shq[threadIdx.x + st]; }
    __syncthreads();
  }
  if (threadIdx.x == 0) {
    float m = shs[0] / M;
    float var = shq[0] / M - m * m;
    stats[f] = m; stats[F + f] = rsqrtf(var + eps);
  }
}
__global__ void k_bn_apply(const float* __restrict__ X, const float* __restrict__ stats,
                           const float* __restrict__ g, const float* __restrict__ b,
                           int M, int F, int relu,
                           float* __restrict__ Y, _Float16* __restrict__ Yh) {
  int i = blockIdx.x * blockDim.x + threadIdx.x;
  if (i >= M * F) return;
  int f = i % F;
  float v = (X[i] - stats[f]) * stats[F + f] * g[f] + b[f];
  if (relu) v = fmaxf(v, 0.f);
  if (Y)  Y[i]  = v;
  if (Yh) Yh[i] = (_Float16)v;
}
// base = relu(bn(gene_emb)) + 0.2*pos  -> f16
__global__ void k_base(const float* __restrict__ ge, const float* __restrict__ stats,
                       const float* __restrict__ g, const float* __restrict__ b,
                       const float* __restrict__ pos, _Float16* __restrict__ out16, int n) {
  int i = blockIdx.x * blockDim.x + threadIdx.x;
  if (i >= n) return;
  int f = i & 63;
  float v = (ge[i] - stats[f]) * stats[64 + f] * g[f] + b[f];
  out16[i] = (_Float16)(fmaxf(v, 0.f) + 0.2f * pos[i]);
}
__global__ void k_psum(const float* __restrict__ pge, const int* __restrict__ pidx,
                       float* __restrict__ psum) {
  int t = threadIdx.x;   // 256
  int b = t >> 6, f = t & 63;
  psum[t] = pge[pidx[b * 2] * 64 + f] + pge[pidx[b * 2 + 1] * 64 + f];
}
__global__ void k_pf_mlp(const float* __restrict__ psum,
                         const float* __restrict__ w1, const float* __restrict__ b1,
                         const float* __restrict__ g1, const float* __restrict__ be1,
                         const float* __restrict__ w2, const float* __restrict__ b2,
                         const float* __restrict__ g2, const float* __restrict__ be2,
                         float* __restrict__ etot) {
  __shared__ float hh[256];
  __shared__ float red[128];
  int t = threadIdx.x; int b = t >> 6, f = t & 63;
  float acc = b1[f];
  for (int k = 0; k < 64; k++) acc += psum[b * 64 + k] * w1[f * 64 + k];
  hh[t] = acc; __syncthreads();
  if (t < 64) {
    float s = 0.f, ss = 0.f;
    for (int i = 0; i < 4; i++) { float v = hh[i * 64 + t]; s += v; ss += v * v; }
    float mm = s * 0.25f; red[t] = mm; red[64 + t] = rsqrtf(ss * 0.25f - mm * mm + 1e-5f);
  }
  __syncthreads();
  float hb = fmaxf((hh[t] - red[f]) * red[64 + f] * g1[f] + be1[f], 0.f);
  __syncthreads(); hh[t] = hb; __syncthreads();
  float acc2 = b2[f];
  for (int k = 0; k < 64; k++) acc2 += hh[b * 64 + k] * w2[f * 64 + k];
  __syncthreads(); hh[t] = acc2; __syncthreads();
  if (t < 64) {
    float s = 0.f, ss = 0.f;
    for (int i = 0; i < 4; i++) { float v = hh[i * 64 + t]; s += v; ss += v * v; }
    float mm = s * 0.25f; red[t] = mm; red[64 + t] = rsqrtf(ss * 0.25f - mm * mm + 1e-5f);
  }
  __syncthreads();
  etot[t] = (hh[t] - red[f]) * red[64 + f] * g2[f] + be2[f];
}
__global__ void k_addb(const float* __restrict__ base, const float* __restrict__ etot,
                       float* __restrict__ flat) {
  int i = blockIdx.x * blockDim.x + threadIdx.x;
  if (i >= B_ * G_ * H_) return;
  int h = i & 63; int row = i >> 6; int g = row & (G_ - 1); int b = row >> 12;
  flat[i] = base[g * 64 + h] + etot[b * 64 + h];
}
__global__ void k_qkv_split(const float* __restrict__ qkv, _Float16* __restrict__ qh,
                            _Float16* __restrict__ kh, _Float16* __restrict__ vh) {
  int i = blockIdx.x * blockDim.x + threadIdx.x;
  if (i >= 2 * B_ * G_ * HD_) return;
  int d = i & 31; int g = (i >> 5) & (G_ - 1); int bh = i >> 17;
  int b = bh >> 1, h = bh & 1;
  size_t src = ((size_t)(b * G_ + g)) * 192 + h * 32 + d;
  qh[i] = (_Float16)qkv[src];
  kh[i] = (_Float16)qkv[src + 64];
  vh[i] = (_Float16)qkv[src + 128];
}
// LayerNorm of (X+R) per 64-wide row; one wave per row (wave32, 2 elems/lane)
__global__ void k_ln(const float* __restrict__ X, const float* __restrict__ R,
                     const float* __restrict__ g, const float* __restrict__ bb,
                     float* __restrict__ Y, _Float16* __restrict__ Yh) {
  int row = blockIdx.x; int lane = threadIdx.x & 31;
  size_t base = (size_t)row * 64;
  float v0 = X[base + lane] + R[base + lane];
  float v1 = X[base + 32 + lane] + R[base + 32 + lane];
  float s = v0 + v1;
  for (int off = 16; off; off >>= 1) s += __shfl_xor(s, off, 32);
  float mean = s * (1.f / 64.f);
  float d0 = v0 - mean, d1 = v1 - mean;
  float ss = d0 * d0 + d1 * d1;
  for (int off = 16; off; off >>= 1) ss += __shfl_xor(ss, off, 32);
  float istd = rsqrtf(ss * (1.f / 64.f) + 1e-5f);
  float y0 = d0 * istd * g[lane] + bb[lane];
  float y1 = d1 * istd * g[lane + 32] + bb[lane + 32];
  Y[base + lane] = y0; Y[base + 32 + lane] = y1;
  if (Yh) { Yh[base + lane] = (_Float16)y0; Yh[base + 32 + lane] = (_Float16)y1; }
}
// he = relu(x @ vae_e1_w.T + b) : one wave per (b, j) output, 4096-dot
__global__ void k_vae_he(const float* __restrict__ x, const float* __restrict__ w,
                         const float* __restrict__ bias, float* __restrict__ he) {
  int o = blockIdx.x; int b = o >> 5, j = o & 31;
  int lane = threadIdx.x & 31;
  const float* xr = x + b * G_;
  const float* wr = w + (size_t)j * G_;
  float s = 0.f;
  for (int k = lane; k < G_; k += 32) s += xr[k] * wr[k];
  for (int off = 16; off; off >>= 1) s += __shfl_xor(s, off, 32);
  if (lane == 0) he[o] = fmaxf(s + bias[j], 0.f);
}
__global__ void k_vae_mid(const float* __restrict__ he,
                          const float* __restrict__ muw, const float* __restrict__ mub,
                          const float* __restrict__ lvw, const float* __restrict__ lvb,
                          const float* __restrict__ d1w, const float* __restrict__ d1b,
                          const float* __restrict__ eps,
                          float* __restrict__ hdc, float* __restrict__ klout) {
  __shared__ float zsh[64];
  __shared__ float klp[64];
  int t = threadIdx.x;  // 128
  if (t < 64) {
    int b = t >> 4, l = t & 15;
    float mm = mub[l], lv = lvb[l];
    for (int i = 0; i < 32; i++) { float h = he[b * 32 + i];
      mm += h * muw[l * 32 + i]; lv += h * lvw[l * 32 + i]; }
    zsh[t] = mm + eps[b * 16 + l] * __expf(0.5f * lv);
    klp[t] = 1.f + lv - mm * mm - __expf(lv);
  }
  __syncthreads();
  { int b = t >> 5, i = t & 31;
    float h = d1b[i];
    for (int l = 0; l < 16; l++) h += zsh[b * 16 + l] * d1w[i * 16 + l];
    hdc[t] = fmaxf(h, 0.f);
  }
  if (t == 0) {
    float s = 0.f; for (int j = 0; j < 64; j++) s += klp[j];
    *klout = -0.5f * (s * 0.25f);
  }
}
// pred = (enc . indv_w1[g]) + indv_b1[g] + where(x==0, 0, hdc . d2_w[g] + d2_b[g])
__global__ void k_final(const float* __restrict__ enc, const float* __restrict__ iw,
                        const float* __restrict__ ib, const float* __restrict__ x,
                        const float* __restrict__ hdc, const float* __restrict__ d2w,
                        const float* __restrict__ d2b, float* __restrict__ outp) {
  int i = blockIdx.x * blockDim.x + threadIdx.x;
  if (i >= B_ * G_) return;
  int g = i & (G_ - 1); int b = i >> 12;
  const float* er = enc + (size_t)i * 64;
  const float* wr = iw + (size_t)g * 64;
  float p = ib[g];
  for (int h = 0; h < 64; h++) p += er[h] * wr[h];
  float rec = d2b[g];
  for (int j = 0; j < 32; j++) rec += hdc[b * 32 + j] * d2w[g * 32 + j];
  if (x[i] == 0.f) rec = 0.f;
  outp[i] = p + rec;
}

// ---------------------------------------------------------------------------
extern "C" void kernel_launch(void* const* d_in, const int* in_sizes, int n_in,
                              void* d_out, int out_size, void* d_ws, size_t ws_size,
                              hipStream_t stream) {
  (void)in_sizes; (void)n_in; (void)out_size; (void)ws_size;
  const int E1 = 81920, E2 = 81920;
  const float* x        = (const float*)d_in[0];
  const int*   pert_idx = (const int*)d_in[1];
  const float* mask     = (const float*)d_in[2];
  const int*   ei_co    = (const int*)d_in[3];
  const float* w_co     = (const float*)d_in[4];
  const int*   ei_go    = (const int*)d_in[5];
  const float* w_go     = (const float*)d_in[6];
  const float* gene_emb = (const float*)d_in[7];
  const float* emb_pos  = (const float*)d_in[8];
  const float* pert_tbl = (const float*)d_in[9];
  const float* sg_co_w  = (const float*)d_in[10];
  const float* sg_co_b  = (const float*)d_in[11];
  const float* sg_go_w  = (const float*)d_in[12];
  const float* sg_go_b  = (const float*)d_in[13];
  const float* bn_emb_g = (const float*)d_in[14];
  const float* bn_emb_b = (const float*)d_in[15];
  const float* e1w = (const float*)d_in[16]; const float* e1b = (const float*)d_in[17];
  const float* e1g = (const float*)d_in[18]; const float* e1be= (const float*)d_in[19];
  const float* e2w = (const float*)d_in[20]; const float* e2b = (const float*)d_in[21];
  const float* e2g = (const float*)d_in[22]; const float* e2be= (const float*)d_in[23];
  const float* p1w = (const float*)d_in[24]; const float* p1b = (const float*)d_in[25];
  const float* p1g = (const float*)d_in[26]; const float* p1be= (const float*)d_in[27];
  const float* p2w = (const float*)d_in[28]; const float* p2b = (const float*)d_in[29];
  const float* p2g = (const float*)d_in[30]; const float* p2be= (const float*)d_in[31];
  const float* r1w = (const float*)d_in[32]; const float* r1b = (const float*)d_in[33];
  const float* r1g = (const float*)d_in[34]; const float* r1be= (const float*)d_in[35];
  const float* r2w = (const float*)d_in[36]; const float* r2b = (const float*)d_in[37];
  const float* r2g = (const float*)d_in[38]; const float* r2be= (const float*)d_in[39];
  const float* bnpb_g = (const float*)d_in[40]; const float* bnpb_b = (const float*)d_in[41];
  const float* ain_w  = (const float*)d_in[42]; const float* ain_b  = (const float*)d_in[43];
  const float* aout_w = (const float*)d_in[44]; const float* aout_b = (const float*)d_in[45];
  const float* ln1_g  = (const float*)d_in[46]; const float* ln1_b  = (const float*)d_in[47];
  const float* ff1_w  = (const float*)d_in[48]; const float* ff1_b  = (const float*)d_in[49];
  const float* ff2_w  = (const float*)d_in[50]; const float* ff2_b  = (const float*)d_in[51];
  const float* ln2_g  = (const float*)d_in[52]; const float* ln2_b  = (const float*)d_in[53];
  const float* indv_w = (const float*)d_in[54]; const float* indv_b = (const float*)d_in[55];
  const float* ve1w = (const float*)d_in[56]; const float* ve1b = (const float*)d_in[57];
  const float* vmuw = (const float*)d_in[58]; const float* vmub = (const float*)d_in[59];
  const float* vlvw = (const float*)d_in[60]; const float* vlvb = (const float*)d_in[61];
  const float* vd1w = (const float*)d_in[62]; const float* vd1b = (const float*)d_in[63];
  const float* vd2w = (const float*)d_in[64]; const float* vd2b = (const float*)d_in[65];
  const float* eps  = (const float*)d_in[66];
  float* outp = (float*)d_out;

  // ---- deterministic bump allocator over d_ws
  char* cur = (char*)d_ws;
  auto allocf = [&](size_t n) { float* p = (float*)cur; cur += ((n * 4 + 255) / 256) * 256; return p; };
  auto alloch = [&](size_t n) { _Float16* p = (_Float16*)cur; cur += ((n * 2 + 255) / 256) * 256; return p; };
  float* deg   = allocf(G_);
  float* dinv  = allocf(G_);
  float* aggA  = allocf(G_ * 64);
  float* pos   = allocf(G_ * 64);
  float* stats = allocf(1024);
  float* tmpA  = allocf(G_ * 64);
  float* baseF = allocf(G_ * 64);
  float* pge   = allocf(G_ * 64);
  float* psum  = allocf(256);
  float* etot  = allocf(256);
  float* flatF = allocf((size_t)B_ * G_ * 64);
  float* rwh   = allocf((size_t)B_ * G_ * 128);
  float* outF  = allocf((size_t)B_ * G_ * 64);
  float* qkv   = allocf((size_t)B_ * G_ * 192);
  float* ctx   = allocf((size_t)B_ * G_ * 64);
  float* att   = allocf((size_t)B_ * G_ * 64);
  float* h1F   = allocf((size_t)B_ * G_ * 64);
  float* ffh   = allocf((size_t)B_ * G_ * 512);
  float* ffo   = allocf((size_t)B_ * G_ * 64);
  float* encF  = allocf((size_t)B_ * G_ * 64);
  float* he    = allocf(128);
  float* hdc   = allocf(128);
  _Float16* X16a = alloch((size_t)B_ * G_ * 512);
  _Float16* X16b = alloch((size_t)B_ * G_ * 512);
  _Float16* W16  = alloch(512 * 512);
  _Float16* qh   = alloch((size_t)2 * B_ * G_ * HD_);
  _Float16* kh   = alloch((size_t)2 * B_ * G_ * HD_);
  _Float16* vh   = alloch((size_t)2 * B_ * G_ * HD_);

  const int BG = B_ * G_;           // 16384
  const int TPB = 256;
  auto blk = [](int n) { return (n + 255) / 256; };

  // ================= SGConv (coexpress) -> pos =================
  k_fill<<<blk(G_), TPB, 0, stream>>>(deg, G_, 1.0f);                 // self loops
  k_edge_deg<<<blk(E1), TPB, 0, stream>>>(ei_co, ei_co + E1, w_co, deg, E1);
  k_dinv<<<blk(G_), TPB, 0, stream>>>(deg, dinv, G_);
  k_sg_self<<<blk(G_ * 64), TPB, 0, stream>>>(emb_pos, dinv, aggA, G_ * 64);
  k_sg_edge<<<blk(E1 * 64), TPB, 0, stream>>>(ei_co, ei_co + E1, w_co, dinv, emb_pos, aggA, E1);
  k_cast16<<<blk(G_ * 64), TPB, 0, stream>>>(aggA, X16a, G_ * 64);
  k_cast16<<<blk(64 * 64), TPB, 0, stream>>>(sg_co_w, W16, 64 * 64);
  k_gemm_wmma<<<dim3(4, G_ / 16), 32, 0, stream>>>(X16a, W16, sg_co_b, pos, nullptr, G_, 64, 64, 0);

  // ================= base = relu(bn(gene_emb)) + 0.2*pos =================
  k_bn_stats<<<64, 256, 0, stream>>>(gene_emb, G_, 64, stats, 1e-5f);
  k_base<<<blk(G_ * 64), TPB, 0, stream>>>(gene_emb, stats, bn_emb_g, bn_emb_b, pos, X16a, G_ * 64);

  // ================= etv2 MLP =================
  k_cast16<<<blk(64 * 64), TPB, 0, stream>>>(e1w, W16, 64 * 64);
  k_gemm_wmma<<<dim3(4, G_ / 16), 32, 0, stream>>>(X16a, W16, e1b, tmpA, nullptr, G_, 64, 64, 0);
  k_bn_stats<<<64, 256, 0, stream>>>(tmpA, G_, 64, stats, 1e-5f);
  k_bn_apply<<<blk(G_ * 64), TPB, 0, stream>>>(tmpA, stats, e1g, e1be, G_, 64, 1, nullptr, X16a);
  k_cast16<<<blk(64 * 64), TPB, 0, stream>>>(e2w, W16, 64 * 64);
  k_gemm_wmma<<<dim3(4, G_ / 16), 32, 0, stream>>>(X16a, W16, e2b, tmpA, nullptr, G_, 64, 64, 0);
  k_bn_stats<<<64, 256, 0, stream>>>(tmpA, G_, 64, stats, 1e-5f);
  k_bn_apply<<<blk(G_ * 64), TPB, 0, stream>>>(tmpA, stats, e2g, e2be, G_, 64, 0, baseF, nullptr);

  // ================= SGConv (GO) -> pge =================
  k_fill<<<blk(G_), TPB, 0, stream>>>(deg, G_, 1.0f);
  k_edge_deg<<<blk(E2), TPB, 0, stream>>>(ei_go, ei_go + E2, w_go, deg, E2);
  k_dinv<<<blk(G_), TPB, 0, stream>>>(deg, dinv, G_);
  k_sg_self<<<blk(G_ * 64), TPB, 0, stream>>>(pert_tbl, dinv, aggA, G_ * 64);
  k_sg_edge<<<blk(E2 * 64), TPB, 0, stream>>>(ei_go, ei_go + E2, w_go, dinv, pert_tbl, aggA, E2);
  k_cast16<<<blk(G_ * 64), TPB, 0, stream>>>(aggA, X16a, G_ * 64);
  k_cast16<<<blk(64 * 64), TPB, 0, stream>>>(sg_go_w, W16, 64 * 64);
  k_gemm_wmma<<<dim3(4, G_ / 16), 32, 0, stream>>>(X16a, W16, sg_go_b, pge, nullptr, G_, 64, 64, 0);

  // ================= pert fusion MLP =================
  k_psum<<<1, 256, 0, stream>>>(pge, pert_idx, psum);
  k_pf_mlp<<<1, 256, 0, stream>>>(psum, p1w, p1b, p1g, p1be, p2w, p2b, p2g, p2be, etot);

  // ================= base_b -> bnpb -> rw MLP =================
  k_addb<<<blk(BG * 64), TPB, 0, stream>>>(baseF, etot, flatF);
  k_bn_stats<<<64, 256, 0, stream>>>(flatF, BG, 64, stats, 1e-5f);
  k_bn_apply<<<blk(BG * 64), TPB, 0, stream>>>(flatF, stats, bnpb_g, bnpb_b, BG, 64, 1, nullptr, X16a);
  k_cast16<<<blk(128 * 64), TPB, 0, stream>>>(r1w, W16, 128 * 64);
  k_gemm_wmma<<<dim3(8, BG / 16), 32, 0, stream>>>(X16a, W16, r1b, rwh, nullptr, BG, 128, 64, 0);
  k_bn_stats<<<128, 256, 0, stream>>>(rwh, BG, 128, stats, 1e-5f);
  k_bn_apply<<<blk(BG * 128), TPB, 0, stream>>>(rwh, stats, r1g, r1be, BG, 128, 1, nullptr, X16b);
  k_cast16<<<blk(64 * 128), TPB, 0, stream>>>(r2w, W16, 64 * 128);
  k_gemm_wmma<<<dim3(4, BG / 16), 32, 0, stream>>>(X16b, W16, r2b, flatF, nullptr, BG, 64, 128, 0);
  k_bn_stats<<<64, 256, 0, stream>>>(flatF, BG, 64, stats, 1e-5f);
  k_bn_apply<<<blk(BG * 64), TPB, 0, stream>>>(flatF, stats, r2g, r2be, BG, 64, 0, outF, X16a);

  // ================= transformer: attention =================
  k_cast16<<<blk(192 * 64), TPB, 0, stream>>>(ain_w, W16, 192 * 64);
  k_gemm_wmma<<<dim3(12, BG / 16), 32, 0, stream>>>(X16a, W16, ain_b, qkv, nullptr, BG, 192, 64, 0);
  k_qkv_split<<<blk(2 * BG * HD_), TPB, 0, stream>>>(qkv, qh, kh, vh);
  k_flash<<<dim3(G_ / 16, 2 * B_), 32, 0, stream>>>(qh, kh, vh, mask, ctx, 0.17677669529663689f);
  k_cast16<<<blk(BG * 64), TPB, 0, stream>>>(ctx, X16a, BG * 64);
  k_cast16<<<blk(64 * 64), TPB, 0, stream>>>(aout_w, W16, 64 * 64);
  k_gemm_wmma<<<dim3(4, BG / 16), 32, 0, stream>>>(X16a, W16, aout_b, att, nullptr, BG, 64, 64, 0);
  k_ln<<<BG, 32, 0, stream>>>(outF, att, ln1_g, ln1_b, h1F, X16a);

  // ================= transformer: FFN =================
  k_cast16<<<blk(512 * 64), TPB, 0, stream>>>(ff1_w, W16, 512 * 64);
  k_gemm_wmma<<<dim3(32, BG / 16), 32, 0, stream>>>(X16a, W16, ff1_b, ffh, X16b, BG, 512, 64, 1);
  k_cast16<<<blk(64 * 512), TPB, 0, stream>>>(ff2_w, W16, 64 * 512);
  k_gemm_wmma<<<dim3(4, BG / 16), 32, 0, stream>>>(X16b, W16, ff2_b, ffo, nullptr, BG, 64, 512, 0);
  k_ln<<<BG, 32, 0, stream>>>(h1F, ffo, ln2_g, ln2_b, encF, nullptr);

  // ================= VAE branch + final combine =================
  k_vae_he<<<128, 32, 0, stream>>>(x, ve1w, ve1b, he);
  k_vae_mid<<<1, 128, 0, stream>>>(he, vmuw, vmub, vlvw, vlvb, vd1w, vd1b, eps, hdc, outp + BG);
  k_final<<<blk(BG), TPB, 0, stream>>>(encF, indv_w, indv_b, x, hdc, vd2w, vd2b, outp);
}
